// AspectSent_45973329936569
// MI455X (gfx1250) — compile-verified, matrix operand order checked by reading, not run on hardware
//
#include <hip/hip_runtime.h>
#include <hip/hip_bf16.h>

// ---------------------------------------------------------------------------
// AspectSent forward (bi-GRU + CRF marginals + classifier) for gfx1250.
// Heavy GEMMs run on v_wmma_f32_16x16x32_f16 (wave32 WMMA); recurrence is
// parallelized over direction x batch-groups (16 persistent workgroups).
// x_t staging uses the CDNA5 async-to-LDS engine (ASYNCcnt), overlapping the
// next step's fetch with the current step's WMMA + gate phases.
// ---------------------------------------------------------------------------

#define BB   128   // batch
#define LL   512   // sequence length
#define DIN  350   // input feature dim
#define KX   352   // DIN padded to multiple of 32
#define HD   256   // per-direction hidden
#define KTOT 608   // KX + HD  (19 * 32)
#define NCOL 1024  // packed gate columns: [r,z fused | xn | hn]
#define KSTEPS 19
#define BG   16    // batch rows per workgroup
#define HFULL 512  // bidirectional hidden (context width)

typedef __attribute__((ext_vector_type(16))) _Float16 v16h;
typedef __attribute__((ext_vector_type(8)))  float    v8f;

#if defined(__gfx1250__) && __has_builtin(__builtin_amdgcn_global_load_async_to_lds_b128) && \
    __has_builtin(__builtin_amdgcn_global_load_async_to_lds_b64) && \
    __has_builtin(__builtin_amdgcn_s_wait_asynccnt)
#define ASYNC_LDS 1
typedef int v4i_t __attribute__((ext_vector_type(4)));
typedef int v2i_t __attribute__((ext_vector_type(2)));
typedef __attribute__((address_space(1))) v4i_t* gptr4_t;  // global int4*
typedef __attribute__((address_space(1))) v2i_t* gptr2_t;  // global int2*
typedef __attribute__((address_space(3))) v4i_t* lptr4_t;  // LDS int4*
typedef __attribute__((address_space(3))) v2i_t* lptr2_t;  // LDS int2*
#else
#define ASYNC_LDS 0
#endif

// K-offset within a 32-wide K slab for the 16-bit WMMA fragment layout
// (ISA 7.12.2): lanes 0-15 hold K {0..7,16..23}, lanes 16-31 hold {8..15,24..31}.
__device__ __forceinline__ int koff(int e, int hi) {
  return (e < 8) ? (hi ? e + 8 : e) : (hi ? e + 16 : e + 8);
}

// ---------------------------------------------------------------------------
// Pack f32 weights into the fused f16 layout + combined bias.
// Wcat[dir][row][k]:
//   row <  512 : k<350 -> w_ih[row][k], 352<=k -> w_hh[row][k-352]   (r,z fused)
//   row <  768 : k<350 -> w_ih[row][k]                               (xn)
//   row < 1024 : 352<=k -> w_hh[row-256][k-352]                      (hn)
// ---------------------------------------------------------------------------
__global__ __launch_bounds__(256) void prep_weights_kernel(
    const float* __restrict__ wihf, const float* __restrict__ whhf,
    const float* __restrict__ bihf, const float* __restrict__ bhhf,
    const float* __restrict__ wihb, const float* __restrict__ whhb,
    const float* __restrict__ bihb, const float* __restrict__ bhhb,
    _Float16* __restrict__ Wcat, float* __restrict__ biasc) {
  int idx = blockIdx.x * 256 + threadIdx.x;
  const int total = 2 * NCOL * KTOT;
  if (idx >= total) return;
  int dir = idx / (NCOL * KTOT);
  int rem = idx - dir * NCOL * KTOT;
  int row = rem / KTOT;
  int k   = rem - row * KTOT;
  const float* wih = dir ? wihb : wihf;
  const float* whh = dir ? whhb : whhf;
  float v = 0.0f;
  if (row < 512) {
    if (k < DIN)      v = wih[row * DIN + k];
    else if (k >= KX) v = whh[row * HD + (k - KX)];
  } else if (row < 768) {
    if (k < DIN)      v = wih[row * DIN + k];
  } else {
    if (k >= KX)      v = whh[(row - 256) * HD + (k - KX)];
  }
  Wcat[idx] = (_Float16)v;
  if (k == 0) {
    const float* bih = dir ? bihb : bihf;
    const float* bhh = dir ? bhhb : bhhf;
    float bv;
    if (row < 512)      bv = bih[row] + bhh[row];
    else if (row < 768) bv = bih[row];
    else                bv = bhh[row - 256];
    biasc[dir * NCOL + row] = bv;
  }
}

// ---------------------------------------------------------------------------
// Persistent bi-GRU recurrence. 16 blocks: dir = bx>>3, batch-group = bx&7.
// Per step: WMMA GEMM [16 x 1024] = A[16 x 608] * B[608 x 1024], then gates.
// Writes context[b][l][dir*256 + j] = wm ? tanh(h) : 0.
// ---------------------------------------------------------------------------
__global__ __launch_bounds__(256) void gru_kernel(
    const float* __restrict__ sents, const int* __restrict__ lens,
    const _Float16* __restrict__ Wcat, const float* __restrict__ biasc,
    float* __restrict__ hgws, float* __restrict__ ctx) {
  const int bx   = blockIdx.x;
  const int dir  = bx >> 3;
  const int grp  = bx & 7;
  const int r0   = grp * BG;
  const int tid  = threadIdx.x;
  const int lane = tid & 31;
  const int wave = tid >> 5;
  const int hi   = (lane >> 4) & 1;
  const int lidx = lane & 15;

  __shared__ _Float16 A_lds[BG][KTOT];   // [x_t f16 | h f16]   19.0 KB
  __shared__ float    h_cur[BG][HD];     //                     16.0 KB
#if ASYNC_LDS
  __shared__ float    xstage[BG][KX];    // raw f32 x_t slab    22.0 KB
#endif

  const _Float16* wc   = Wcat  + (size_t)dir * NCOL * KTOT;
  const float*    bias = biasc + dir * NCOL;
  float*          hg   = hgws  + (size_t)bx * BG * NCOL;

  const v8f vzero = {0.f, 0.f, 0.f, 0.f, 0.f, 0.f, 0.f, 0.f};

  for (int i = tid; i < BG * HD; i += 256) h_cur[i / HD][i % HD] = 0.0f;

#if ASYNC_LDS
  // Async fetch of one step's x slab: 87 x b128 + 1 x b64 per row (350 f32).
  auto issue_x = [&](int lpos) {
    for (int i = tid; i < BG * 87; i += 256) {
      int m = i / 87, c = i - m * 87;
      const float* g = sents + ((size_t)(r0 + m) * LL + lpos) * DIN + c * 4;
      __builtin_amdgcn_global_load_async_to_lds_b128(
          (gptr4_t)g, (lptr4_t)&xstage[m][c * 4], 0, 0);
    }
    for (int i = tid; i < BG; i += 256) {
      const float* g = sents + ((size_t)(r0 + i) * LL + lpos) * DIN + 348;
      __builtin_amdgcn_global_load_async_to_lds_b64(
          (gptr2_t)g, (lptr2_t)&xstage[i][348], 0, 0);
    }
  };
  issue_x(dir ? (LL - 1) : 0);  // prologue: fetch step 0's slab
#endif
  __syncthreads();

  for (int t = 0; t < LL; ++t) {
    const int lpos = dir ? (LL - 1 - t) : t;

#if ASYNC_LDS
    // Wait for this step's async slab, then convert f32 -> f16 fragment store.
    __builtin_amdgcn_s_wait_asynccnt(0);
    __syncthreads();
    for (int i = tid; i < BG * KX; i += 256) {
      int m = i / KX, d = i - m * KX;
      A_lds[m][d] = (_Float16)((d < DIN) ? xstage[m][d] : 0.0f);
    }
#else
    // Stage x_t (f32 -> f16, zero-padded K 350..351)
    for (int i = tid; i < BG * KX; i += 256) {
      int m = i / KX, d = i - m * KX;
      float v = (d < DIN) ? sents[((size_t)(r0 + m) * LL + lpos) * DIN + d] : 0.0f;
      A_lds[m][d] = (_Float16)v;
    }
#endif
    // Stage h (f32 -> f16)
    for (int i = tid; i < BG * HD; i += 256) {
      int m = i / HD, j = i - m * HD;
      A_lds[m][KX + j] = (_Float16)h_cur[m][j];
    }
    __syncthreads();

#if ASYNC_LDS
    // xstage consumed; overlap next step's fetch with WMMA + gate phases.
    if (t + 1 < LL) issue_x(dir ? (LL - 2 - t) : (t + 1));
#endif

    // WMMA phase: 64 N-tiles of 16x16, 8 per wave, K-loop of 19 x 32.
    v8f acc[8];
#pragma unroll
    for (int i = 0; i < 8; ++i) acc[i] = vzero;

    for (int ks = 0; ks < KSTEPS; ++ks) {
      const int kb = ks * 32;
      v16h afrag;
#pragma unroll
      for (int e = 0; e < 16; ++e) afrag[e] = A_lds[lidx][kb + koff(e, hi)];
      // keep the weight stream flowing: global_prefetch_b8 for next K slab
      if (ks + 1 < KSTEPS)
        __builtin_prefetch(wc + (size_t)((wave * 8) * 16 + lidx) * KTOT + kb + 32, 0, 1);
#pragma unroll
      for (int i = 0; i < 8; ++i) {
        const int n = (wave * 8 + i) * 16 + lidx;
        const _Float16* wrow = wc + (size_t)n * KTOT + kb;
        v16h bfrag;
#pragma unroll
        for (int e = 0; e < 16; ++e) bfrag[e] = wrow[koff(e, hi)];
        acc[i] = __builtin_amdgcn_wmma_f32_16x16x32_f16(
            false, afrag, false, bfrag, (short)0, acc[i], false, false);
      }
    }

    // Epilogue: C VGPR r holds row r (+8 for hi lanes), col = tile*16 + lidx.
#pragma unroll
    for (int i = 0; i < 8; ++i) {
      const int col = (wave * 8 + i) * 16 + lidx;
      const float bv = bias[col];
#pragma unroll
      for (int r = 0; r < 8; ++r) {
        const int m = r + (hi ? 8 : 0);
        hg[m * NCOL + col] = acc[i][r] + bv;
      }
    }
    __threadfence_block();
    __syncthreads();

    // Gate phase: r,z fused preacts in cols [0,512); xn in [512,768); hn in [768,1024)
    for (int i = tid; i < BG * HD; i += 256) {
      int m = i / HD, j = i - m * HD;
      int bglob = r0 + m;
      float pr = hg[m * NCOL + j];
      float pz = hg[m * NCOL + 256 + j];
      float xn = hg[m * NCOL + 512 + j];
      float hn = hg[m * NCOL + 768 + j];
      float rv = 1.0f / (1.0f + expf(-pr));
      float zv = 1.0f / (1.0f + expf(-pz));
      float nv = tanhf(xn + rv * hn);
      float hold = h_cur[m][j];
      float hnew = (1.0f - zv) * nv + zv * hold;
      bool valid = lpos < lens[bglob];
      float hc = valid ? hnew : hold;   // packed-sequence masked carry
      h_cur[m][j] = hc;
      float outv = valid ? tanhf(hc) : 0.0f;  // hs * wm, then tanh(concat)
      ctx[((size_t)bglob * LL + lpos) * HFULL + (dir ? HD : 0) + j] = outv;
    }
    __syncthreads();
  }
}

// ---------------------------------------------------------------------------
// Target-average over masked tokens: tavg[b][h] = sum_l m*ctx / sum_l m
// ---------------------------------------------------------------------------
__global__ __launch_bounds__(256) void tgt_avg_kernel(
    const float* __restrict__ ctx, const int* __restrict__ masks,
    float* __restrict__ tavg) {
  int b = blockIdx.x, tid = threadIdx.x;
  float msum = 0.0f;
  for (int l = 0; l < LL; ++l) msum += (float)masks[b * LL + l];
  for (int h = tid; h < HFULL; h += 256) {
    float s = 0.0f;
    for (int l = 0; l < LL; ++l)
      if (masks[b * LL + l]) s += ctx[((size_t)b * LL + l) * HFULL + h];
    tavg[b * HFULL + h] = s / msum;
  }
}

__global__ __launch_bounds__(256) void add_avg_kernel(
    float* __restrict__ ctx, const float* __restrict__ tavg) {
  size_t idx = (size_t)blockIdx.x * 256 + threadIdx.x;  // B*L*HFULL exact
  int h = (int)(idx & (HFULL - 1));
  int b = (int)(idx >> 9) / LL;
  ctx[idx] += tavg[b * HFULL + h];
}

// feats[b][l][tg] = ctx[b][l] . w_tri[tg] + b_tri[tg]
__global__ __launch_bounds__(256) void feats_kernel(
    const float* __restrict__ ctx, const float* __restrict__ w_tri,
    const float* __restrict__ b_tri, float* __restrict__ feats) {
  int idx = blockIdx.x * 256 + threadIdx.x;  // B*L*4
  int tg = idx & 3;
  int bl = idx >> 2;
  const float* c = ctx + (size_t)bl * HFULL;
  const float* w = w_tri + tg * HFULL;
  float s = b_tri[tg];
  for (int h = 0; h < HFULL; ++h) s += c[h] * w[h];
  feats[idx] = s;
}

// ---------------------------------------------------------------------------
// 4-state CRF forward/backward (one thread per sample, masked carry).
// ---------------------------------------------------------------------------
__global__ __launch_bounds__(128) void crf_kernel(
    const float* __restrict__ feats, const float* __restrict__ trans,
    const int* __restrict__ lens, float* __restrict__ alphas,
    float* __restrict__ betas) {
  int b = threadIdx.x;
  float tr[16];
#pragma unroll
  for (int i = 0; i < 16; ++i) tr[i] = trans[i];
  int len = lens[b];
  const float* fb = feats + (size_t)b * LL * 4;

  float a[4];
#pragma unroll
  for (int t = 0; t < 4; ++t) {
    a[t] = fb[t];
    alphas[(size_t)b * 4 + t] = a[t];
  }
  for (int l = 1; l < LL; ++l) {
    if (l < len) {
      float an[4];
#pragma unroll
      for (int j = 0; j < 4; ++j) {
        float v0 = a[0] + tr[0 * 4 + j], v1 = a[1] + tr[1 * 4 + j];
        float v2 = a[2] + tr[2 * 4 + j], v3 = a[3] + tr[3 * 4 + j];
        float mx = fmaxf(fmaxf(v0, v1), fmaxf(v2, v3));
        an[j] = mx + logf(expf(v0 - mx) + expf(v1 - mx) + expf(v2 - mx) + expf(v3 - mx))
                + fb[l * 4 + j];
      }
#pragma unroll
      for (int t = 0; t < 4; ++t) a[t] = an[t];
    }
#pragma unroll
    for (int t = 0; t < 4; ++t) alphas[(size_t)l * BB * 4 + b * 4 + t] = a[t];
  }

  float bv[4] = {0.f, 0.f, 0.f, 0.f};
#pragma unroll
  for (int t = 0; t < 4; ++t) betas[(size_t)(LL - 1) * BB * 4 + b * 4 + t] = 0.0f;
  for (int l = LL - 2; l >= 0; --l) {
    if (l + 1 < len) {
      float s0 = fb[(l + 1) * 4 + 0] + bv[0], s1 = fb[(l + 1) * 4 + 1] + bv[1];
      float s2 = fb[(l + 1) * 4 + 2] + bv[2], s3 = fb[(l + 1) * 4 + 3] + bv[3];
      float bn[4];
#pragma unroll
      for (int i = 0; i < 4; ++i) {
        float v0 = tr[i * 4 + 0] + s0, v1 = tr[i * 4 + 1] + s1;
        float v2 = tr[i * 4 + 2] + s2, v3 = tr[i * 4 + 3] + s3;
        float mx = fmaxf(fmaxf(v0, v1), fmaxf(v2, v3));
        bn[i] = mx + logf(expf(v0 - mx) + expf(v1 - mx) + expf(v2 - mx) + expf(v3 - mx));
      }
#pragma unroll
      for (int t = 0; t < 4; ++t) bv[t] = bn[t];
    }
#pragma unroll
    for (int t = 0; t < 4; ++t) betas[(size_t)l * BB * 4 + b * 4 + t] = bv[t];
  }
}

// p[b][l] = softmax(alpha+beta)[1] * wm
__global__ __launch_bounds__(256) void marg_kernel(
    const float* __restrict__ alphas, const float* __restrict__ betas,
    const int* __restrict__ lens, float* __restrict__ p) {
  int idx = blockIdx.x * 256 + threadIdx.x;  // B*L
  int b = idx / LL, l = idx - b * LL;
  size_t o = (size_t)l * BB * 4 + b * 4;
  float v0 = alphas[o + 0] + betas[o + 0];
  float v1 = alphas[o + 1] + betas[o + 1];
  float v2 = alphas[o + 2] + betas[o + 2];
  float v3 = alphas[o + 3] + betas[o + 3];
  float mx = fmaxf(fmaxf(v0, v1), fmaxf(v2, v3));
  float s = expf(v0 - mx) + expf(v1 - mx) + expf(v2 - mx) + expf(v3 - mx);
  float pv = expf(v1 - mx) / s;
  p[idx] = (l < lens[b]) ? pv : 0.0f;
}

// gamma/sent_v/L1 per sample
__global__ __launch_bounds__(256) void pool_kernel(
    const float* __restrict__ p, const float* __restrict__ ctx,
    float* __restrict__ sent_v, float* __restrict__ pl1) {
  int b = blockIdx.x, tid = threadIdx.x;
  __shared__ float red[256];
  float la = 0.0f, ls = 0.0f;
  for (int l = tid; l < LL; l += 256) {
    float pv = p[b * LL + l];
    la += fabsf(pv);
    ls += pv;
  }
  red[tid] = la;
  __syncthreads();
  for (int s = 128; s > 0; s >>= 1) { if (tid < s) red[tid] += red[tid + s]; __syncthreads(); }
  float sumabs = red[0];
  __syncthreads();
  red[tid] = ls;
  __syncthreads();
  for (int s = 128; s > 0; s >>= 1) { if (tid < s) red[tid] += red[tid + s]; __syncthreads(); }
  float gamma = red[0] * 0.5f;
  if (tid == 0) pl1[b] = sumabs;
  for (int h = tid; h < HFULL; h += 256) {
    float s = 0.0f;
    for (int l = 0; l < LL; ++l)
      s += p[b * LL + l] * ctx[((size_t)b * LL + l) * HFULL + h];
    sent_v[b * HFULL + h] = s / gamma;
  }
}

// label scores -> cls_loss; pena + mean L1 -> norm_pen
__global__ __launch_bounds__(128) void final_kernel(
    const float* __restrict__ sent_v, const float* __restrict__ w_lab,
    const float* __restrict__ b_lab, const int* __restrict__ labels,
    const float* __restrict__ trans, const float* __restrict__ pl1,
    float* __restrict__ out) {
  int b = threadIdx.x;
  __shared__ float red[128];
  float sc[3];
#pragma unroll
  for (int c = 0; c < 3; ++c) {
    float s = b_lab[c];
    const float* w = w_lab + c * HFULL;
    const float* v = sent_v + b * HFULL;
    for (int h = 0; h < HFULL; ++h) s += v[h] * w[h];
    sc[c] = s;
  }
  float mx = fmaxf(sc[0], fmaxf(sc[1], sc[2]));
  float lse = mx + logf(expf(sc[0] - mx) + expf(sc[1] - mx) + expf(sc[2] - mx));
  int li = labels[b];
  float lossb = lse - sc[li];
  red[b] = lossb;
  __syncthreads();
  for (int s = 64; s > 0; s >>= 1) { if (b < s) red[b] += red[b + s]; __syncthreads(); }
  float cls = red[0] / (float)BB;
  __syncthreads();
  red[b] = pl1[b];
  __syncthreads();
  for (int s = 64; s > 0; s >>= 1) { if (b < s) red[b] += red[b + s]; __syncthreads(); }
  if (b == 0) {
    float norm = red[0] / (float)BB;
    float pena = fmaxf(0.0f, trans[4] - trans[0]) + fmaxf(0.0f, trans[1] - trans[5]);
    out[0] = cls;
    out[1] = 1.0f * pena + 1.0f * norm;  // C1 = C2 = 1
  }
}

// ---------------------------------------------------------------------------
extern "C" void kernel_launch(void* const* d_in, const int* in_sizes, int n_in,
                              void* d_out, int out_size, void* d_ws, size_t ws_size,
                              hipStream_t stream) {
  const float* sents  = (const float*)d_in[0];
  const int*   masks  = (const int*)  d_in[1];
  const int*   labels = (const int*)  d_in[2];
  const int*   lens   = (const int*)  d_in[3];
  const float* w_ih_f = (const float*)d_in[4];
  const float* w_hh_f = (const float*)d_in[5];
  const float* b_ih_f = (const float*)d_in[6];
  const float* b_hh_f = (const float*)d_in[7];
  const float* w_ih_b = (const float*)d_in[8];
  const float* w_hh_b = (const float*)d_in[9];
  const float* b_ih_b = (const float*)d_in[10];
  const float* b_hh_b = (const float*)d_in[11];
  const float* w_tri  = (const float*)d_in[12];
  const float* b_tri  = (const float*)d_in[13];
  const float* trans  = (const float*)d_in[14];
  const float* w_lab  = (const float*)d_in[15];
  const float* b_lab  = (const float*)d_in[16];
  float* out = (float*)d_out;

  // Workspace carve-out (~137 MB total); everything read is written each call.
  char* ws = (char*)d_ws;
  size_t off = 0;
  auto take = [&](size_t bytes) -> char* {
    char* ptr = ws + off;
    off += (bytes + 255) & ~(size_t)255;
    return ptr;
  };
  _Float16* Wcat  = (_Float16*)take((size_t)2 * NCOL * KTOT * sizeof(_Float16));
  float*    biasc = (float*)take((size_t)2 * NCOL * sizeof(float));
  float*    hgws  = (float*)take((size_t)16 * BG * NCOL * sizeof(float));
  float*    ctx   = (float*)take((size_t)BB * LL * HFULL * sizeof(float));
  float*    tavg  = (float*)take((size_t)BB * HFULL * sizeof(float));
  float*    feats = (float*)take((size_t)BB * LL * 4 * sizeof(float));
  float*    alphas= (float*)take((size_t)LL * BB * 4 * sizeof(float));
  float*    betas = (float*)take((size_t)LL * BB * 4 * sizeof(float));
  float*    pbuf  = (float*)take((size_t)BB * LL * sizeof(float));
  float*    pl1   = (float*)take((size_t)BB * sizeof(float));
  float*    sentv = (float*)take((size_t)BB * HFULL * sizeof(float));
  (void)ws_size; (void)in_sizes; (void)n_in; (void)out_size;

  {
    int total = 2 * NCOL * KTOT;
    prep_weights_kernel<<<(total + 255) / 256, 256, 0, stream>>>(
        w_ih_f, w_hh_f, b_ih_f, b_hh_f, w_ih_b, w_hh_b, b_ih_b, b_hh_b,
        Wcat, biasc);
  }
  gru_kernel<<<16, 256, 0, stream>>>(sents, lens, Wcat, biasc, hgws, ctx);
  tgt_avg_kernel<<<BB, 256, 0, stream>>>(ctx, masks, tavg);
  {
    size_t total = (size_t)BB * LL * HFULL;
    add_avg_kernel<<<(unsigned)(total / 256), 256, 0, stream>>>(ctx, tavg);
  }
  feats_kernel<<<(BB * LL * 4) / 256, 256, 0, stream>>>(ctx, w_tri, b_tri, feats);
  crf_kernel<<<1, 128, 0, stream>>>(feats, trans, lens, alphas, betas);
  marg_kernel<<<(BB * LL) / 256, 256, 0, stream>>>(alphas, betas, lens, pbuf);
  pool_kernel<<<BB, 256, 0, stream>>>(pbuf, ctx, sentv, pl1);
  final_kernel<<<1, 128, 0, stream>>>(sentv, w_lab, b_lab, labels, trans, pl1, out);
}
